// SpatialAttention2D_75436805587719
// MI455X (gfx1250) — matrix operand order repaired and checked by reading, not compiled, not run on hardware
//
#include <hip/hip_runtime.h>
#include <hip/hip_bf16.h>

typedef __attribute__((ext_vector_type(16))) _Float16 v16h;
typedef __attribute__((ext_vector_type(8)))  float    v8f;

#define BTD   8      // B*T
#define NTOK  1024   // tokens per image
#define CDIM  768
#define NH    12
#define HDIM  64
#define ROWS  (BTD * NTOK)          // 8192
#define QKVN  (3 * CDIM)            // 2304

// ---------------------------------------------------------------------------
// helpers
// ---------------------------------------------------------------------------
__device__ __forceinline__ v16h load_frag16(const _Float16* p0, const _Float16* p1) {
  union { uint4 u[2]; v16h h; } r;
  r.u[0] = *reinterpret_cast<const uint4*>(p0);
  r.u[1] = *reinterpret_cast<const uint4*>(p1);
  return r.h;
}

// ---------------------------------------------------------------------------
// 0) weight convert + transpose:  w_qkv [C][3C] -> wqkvT [3C][C] (f16)
//                                 w_proj [C][C] -> wprojT [C][C] (f16)
// ---------------------------------------------------------------------------
__global__ void prep_weights_kernel(const float* __restrict__ w_qkv,
                                    const float* __restrict__ w_proj,
                                    _Float16* __restrict__ wqkvT,
                                    _Float16* __restrict__ wprojT) {
  const int idx = blockIdx.x * blockDim.x + threadIdx.x;
  const int NQ = QKVN * CDIM;
  if (idx < NQ) {
    const int j = idx / CDIM;         // output col (0..2303)
    const int k = idx - j * CDIM;     // reduction dim
    wqkvT[(size_t)j * CDIM + k] = (_Float16)w_qkv[(size_t)k * QKVN + j];
  } else {
    const int i2 = idx - NQ;          // over 768*768
    const int j = i2 / CDIM;
    const int k = i2 - j * CDIM;
    wprojT[(size_t)j * CDIM + k] = (_Float16)w_proj[(size_t)k * CDIM + j];
  }
}

// ---------------------------------------------------------------------------
// 1) LayerNorm over channel dim, f32 -> f16.  One block (256 thr) per row.
// ---------------------------------------------------------------------------
__global__ void ln_kernel(const float* __restrict__ x,
                          const float* __restrict__ gamma,
                          const float* __restrict__ beta,
                          _Float16* __restrict__ xn) {
  const int row = blockIdx.x;
  const int t = threadIdx.x;
  const float* xr = x + (size_t)row * CDIM;
  float v[3];
  float s = 0.f, ss = 0.f;
#pragma unroll
  for (int i = 0; i < 3; ++i) {
    v[i] = xr[t + i * 256];
    s += v[i];
    ss += v[i] * v[i];
  }
  __shared__ float r1[256], r2[256];
  r1[t] = s; r2[t] = ss;
  __syncthreads();
  for (int off = 128; off > 0; off >>= 1) {
    if (t < off) { r1[t] += r1[t + off]; r2[t] += r2[t + off]; }
    __syncthreads();
  }
  const float mu   = r1[0] * (1.0f / CDIM);
  const float var  = r2[0] * (1.0f / CDIM) - mu * mu;
  const float rstd = rsqrtf(var + 1e-5f);
  _Float16* xo = xn + (size_t)row * CDIM;
#pragma unroll
  for (int i = 0; i < 3; ++i) {
    const int c = t + i * 256;
    xo[c] = (_Float16)((v[i] - mu) * rstd * gamma[c] + beta[c]);
  }
}

// ---------------------------------------------------------------------------
// 2) WMMA GEMM:  C[M][N] = A[M][K] * Bt[N][K]^T + bias[N]
//    block = 128 threads (4 waves); each wave computes a 64x64 tile
//    (4x4 grid of 16x16 WMMA tiles -> 16 WMMAs per k-step on 8 fragment
//    loads).  __launch_bounds__(128,1) unlocks the full VGPR file so the
//    128 accumulator VGPRs + fragments stay in registers (no scratch).
//    A-fragment (16x32 f16): lane (g=l>>4,h=l&15): row m+h,
//      halves { k0+8g .. +7 , k0+16+8g .. +7 }  -> two b128 loads
//    B-fragment (32x16 f16): lane: col n+h, halves { k0+8g .. +7 , k0+16+8g .. +7 } -> two b128
//    C (16x16 f32): lane: n = h, m = 8g + r (vgpr r)
// ---------------------------------------------------------------------------
template <bool OUT_F32>
__global__ __launch_bounds__(128, 1)
void gemm_f16_wmma(const _Float16* __restrict__ A,
                   const _Float16* __restrict__ Bt,
                   const float* __restrict__ bias,
                   void* __restrict__ Cv,
                   int M, int N, int K) {
  const int lane = threadIdx.x & 31;
  const int wave = threadIdx.x >> 5;
  const int g   = lane >> 4;
  const int h16 = lane & 15;
  const int m0 = (blockIdx.x * 4 + wave) * 64;
  const int n0 = blockIdx.y * 64;

  v8f acc[4][4] = {};   // [m-tile][n-tile]
  const _Float16* arow = A + (size_t)(m0 + h16) * K + g * 8;   // + mt*16*K + k0
  const _Float16* brow = Bt + (size_t)(n0 + h16) * K + g * 8; // + nt*16*K + k0
  const size_t rstride = (size_t)16 * K;

  for (int k0 = 0; k0 < K; k0 += 32) {
    if (k0 + 64 < K) {   // pull lines 2 k-steps ahead into WGP$
      __builtin_prefetch(arow + k0 + 64, 0, 3);
      __builtin_prefetch(brow + k0 + 64, 0, 3);
    }
    v16h af[4];
#pragma unroll
    for (int mt = 0; mt < 4; ++mt) {
      const _Float16* ap = arow + (size_t)mt * rstride + k0;
      af[mt] = load_frag16(ap, ap + 16);
    }
#pragma unroll
    for (int nt = 0; nt < 4; ++nt) {
      const _Float16* bp = brow + (size_t)nt * rstride + k0;
      const v16h bf = load_frag16(bp, bp + 16);
#pragma unroll
      for (int mt = 0; mt < 4; ++mt)
        acc[mt][nt] = __builtin_amdgcn_wmma_f32_16x16x32_f16(
            false, af[mt], false, bf, (short)0, acc[mt][nt], false, false);
    }
  }

#pragma unroll
  for (int nt = 0; nt < 4; ++nt) {
    const int n = n0 + nt * 16 + h16;
    const float bv = bias[n];
#pragma unroll
    for (int mt = 0; mt < 4; ++mt) {
#pragma unroll
      for (int r = 0; r < 8; ++r) {
        const int m = m0 + mt * 16 + g * 8 + r;
        const float vv = acc[mt][nt][r] + bv;
        if constexpr (OUT_F32)
          ((float*)Cv)[(size_t)m * N + n] = vv;
        else
          ((_Float16*)Cv)[(size_t)m * N + n] = (_Float16)vv;
      }
    }
  }
}

// ---------------------------------------------------------------------------
// 3) RoPE + repack: qkv [ROWS][2304] f16 -> Q,K [bt][h][n][d], Vt [bt][h][d][n]
//    2D RoPE, GPT-NeoX rotate_half, 32x32 grid, quarter=16.
// ---------------------------------------------------------------------------
__global__ void rope_repack_kernel(const _Float16* __restrict__ qkv,
                                   _Float16* __restrict__ Q,
                                   _Float16* __restrict__ K,
                                   _Float16* __restrict__ Vt) {
  const size_t idx = (size_t)blockIdx.x * blockDim.x + threadIdx.x; // < ROWS*CDIM
  const int d = (int)(idx & 63);
  const int h = (int)((idx >> 6) % NH);
  const size_t row = idx / CDIM;
  const int n  = (int)(row % NTOK);
  const int bt = (int)(row / NTOK);

  const _Float16* base = qkv + row * QKVN;
  const int col = h * HDIM + d;
  const int dp  = (d < 32) ? d + 32 : d - 32;
  const int colp = h * HDIM + dp;

  const float qv = (float)base[col];
  const float kv = (float)base[CDIM + col];
  const float vv = (float)base[2 * CDIM + col];
  const float qp = (float)base[colp];
  const float kp = (float)base[CDIM + colp];
  const float sgn = (d < 32) ? -1.0f : 1.0f;

  // emb[d]: dh = d % 32;  dh<16 -> h_idx * invf[dh] ; else w_idx * invf[dh-16]
  const int dh = d & 31;
  const float jj = (float)(dh & 15);
  const float invf = __expf(-jj * 0.5756462732485115f);  // ln(10000)/16
  const float pos = (dh < 16) ? (float)(n >> 5) : (float)(n & 31);
  float c, s;
  __sincosf(pos * invf, &s, &c);

  const float qo = qv * c + sgn * qp * s;
  const float ko = kv * c + sgn * kp * s;

  const size_t hb = ((size_t)bt * NH + h) * (size_t)(NTOK * HDIM);
  Q[hb + (size_t)n * HDIM + d] = (_Float16)qo;
  K[hb + (size_t)n * HDIM + d] = (_Float16)ko;
  Vt[hb + (size_t)d * NTOK + n] = (_Float16)vv;
}

// ---------------------------------------------------------------------------
// 4) Attention, flash-style with WMMA, S^T formulation (P fragment is a pure
//    per-lane register repack).  Each wave owns 32 query rows (2 row tiles)
//    so K/V fragments are reused twice: 16 WMMAs per 32-key chunk on 8
//    fragment loads.
//    grid = (8 qblocks, 12 heads, 8 bt); block = 128 (4 waves, 32 q rows/wave)
// ---------------------------------------------------------------------------
__global__ __launch_bounds__(128, 1)
void attn_kernel(const _Float16* __restrict__ Qm,
                 const _Float16* __restrict__ Km,
                 const _Float16* __restrict__ Vt,
                 _Float16* __restrict__ Oout) {
  const int lane = threadIdx.x & 31;
  const int wave = threadIdx.x >> 5;
  const int g   = lane >> 4;
  const int h16 = lane & 15;
  const int head = blockIdx.y;
  const int bt   = blockIdx.z;
  const int q0 = blockIdx.x * 128 + wave * 32;
  const size_t base = ((size_t)bt * NH + head) * (size_t)(NTOK * HDIM);
  const _Float16* q  = Qm + base;   // [N][64]
  const _Float16* k  = Km + base;   // [N][64]
  const _Float16* vt = Vt + base;   // [64][N]

  // Q as B operand of S^T = K * Q^T : lane holds qrow = h16, dims contiguous.
  v16h qfrag[2][2];                 // [row-tile][k-step]
#pragma unroll
  for (int rt = 0; rt < 2; ++rt) {
    const _Float16* qp = q + (size_t)(q0 + rt * 16 + h16) * HDIM + g * 8;
    qfrag[rt][0] = load_frag16(qp, qp + 16);
    qfrag[rt][1] = load_frag16(qp + 32, qp + 48);
  }

  v8f o[2][4] = {};                 // [row-tile][d-tile 0,16,32,48]
  float rmax[2] = {-3.0e38f, -3.0e38f};
  float rsum[2] = {0.0f, 0.0f};
  const float scale = 0.125f;       // 1/sqrt(64)

  for (int j = 0; j < NTOK; j += 32) {
    // K fragments, shared by both row tiles: kf[k-step][key-tile]
    v16h kf[2][2];
#pragma unroll
    for (int ks = 0; ks < 2; ++ks) {
      const _Float16* kp0 = k + (size_t)(j + h16) * HDIM + ks * 32 + g * 8;
      kf[ks][0] = load_frag16(kp0, kp0 + 16);
      const _Float16* kp1 = kp0 + 16 * HDIM;
      kf[ks][1] = load_frag16(kp1, kp1 + 16);
    }

    v16h pfrag[2];
    float alpha2[2];
#pragma unroll
    for (int rt = 0; rt < 2; ++rt) {
      // S^T tiles: st0 -> keys j..j+15 (M), st1 -> keys j+16..j+31; N = q rows
      v8f st0 = {}, st1 = {};
#pragma unroll
      for (int ks = 0; ks < 2; ++ks) {
        st0 = __builtin_amdgcn_wmma_f32_16x16x32_f16(
            false, kf[ks][0], false, qfrag[rt][ks], (short)0, st0, false, false);
        st1 = __builtin_amdgcn_wmma_f32_16x16x32_f16(
            false, kf[ks][1], false, qfrag[rt][ks], (short)0, st1, false, false);
      }
      // Online softmax.  Lane holds, for qrow = h16, keys {8g..8g+7} (st0)
      // and {16+8g..+7} (st1); partner lane (l^16) holds the other 16 keys.
      float s0[8], s1[8];
      float cmax = -3.0e38f;
#pragma unroll
      for (int r = 0; r < 8; ++r) {
        s0[r] = st0[r] * scale;
        s1[r] = st1[r] * scale;
        cmax = fmaxf(cmax, fmaxf(s0[r], s1[r]));
      }
      cmax = fmaxf(cmax, __shfl_xor(cmax, 16));
      const float nmax = fmaxf(rmax[rt], cmax);
      alpha2[rt] = __expf(rmax[rt] - nmax);
      float psum = 0.0f;
#pragma unroll
      for (int r = 0; r < 8; ++r) {
        const float p0 = __expf(s0[r] - nmax);
        const float p1 = __expf(s1[r] - nmax);
        psum += p0 + p1;
        pfrag[rt][r]     = (_Float16)p0;
        pfrag[rt][8 + r] = (_Float16)p1;
      }
      psum += __shfl_xor(psum, 16);
      rsum[rt] = rsum[rt] * alpha2[rt] + psum;
      rmax[rt] = nmax;
    }

    // Broadcast per-row alpha to C-layout rows (m = 8g + r lives at lane m).
#pragma unroll
    for (int rt = 0; rt < 2; ++rt) {
      float as8[8];
#pragma unroll
      for (int r = 0; r < 8; ++r) as8[r] = __shfl(alpha2[rt], g * 8 + r);
#pragma unroll
      for (int t = 0; t < 4; ++t)
#pragma unroll
        for (int r = 0; r < 8; ++r) o[rt][t][r] *= as8[r];
    }

    // O += P * V : V fragment per d-tile loaded once, used by both row tiles.
#pragma unroll
    for (int t = 0; t < 4; ++t) {
      const _Float16* vp = vt + (size_t)(t * 16 + h16) * NTOK + j + g * 8;
      const v16h vf = load_frag16(vp, vp + 16);
      o[0][t] = __builtin_amdgcn_wmma_f32_16x16x32_f16(
          false, pfrag[0], false, vf, (short)0, o[0][t], false, false);
      o[1][t] = __builtin_amdgcn_wmma_f32_16x16x32_f16(
          false, pfrag[1], false, vf, (short)0, o[1][t], false, false);
    }
  }

  const int row0 = bt * NTOK;
#pragma unroll
  for (int rt = 0; rt < 2; ++rt) {
    float inv8[8];
#pragma unroll
    for (int r = 0; r < 8; ++r) inv8[r] = 1.0f / __shfl(rsum[rt], g * 8 + r);
#pragma unroll
    for (int r = 0; r < 8; ++r) {
      const int qi = q0 + rt * 16 + g * 8 + r;
      _Float16* orow = Oout + (size_t)(row0 + qi) * CDIM + head * HDIM + h16;
      orow[0]  = (_Float16)(o[rt][0][r] * inv8[r]);
      orow[16] = (_Float16)(o[rt][1][r] * inv8[r]);
      orow[32] = (_Float16)(o[rt][2][r] * inv8[r]);
      orow[48] = (_Float16)(o[rt][3][r] * inv8[r]);
    }
  }
}

// ---------------------------------------------------------------------------
// launcher
// ---------------------------------------------------------------------------
extern "C" void kernel_launch(void* const* d_in, const int* in_sizes, int n_in,
                              void* d_out, int out_size, void* d_ws, size_t ws_size,
                              hipStream_t stream) {
  const float* x      = (const float*)d_in[0];
  const float* ln_g   = (const float*)d_in[1];
  const float* ln_b   = (const float*)d_in[2];
  const float* w_qkv  = (const float*)d_in[3];
  const float* b_qkv  = (const float*)d_in[4];
  const float* w_proj = (const float*)d_in[5];
  const float* b_proj = (const float*)d_in[6];
  float* out = (float*)d_out;

  char* ws = (char*)d_ws;
  size_t off = 0;
  auto carve = [&](size_t bytes) -> char* {
    char* p = ws + off;
    off = (off + bytes + 255) & ~(size_t)255;
    return p;
  };
  _Float16* xn      = (_Float16*)carve((size_t)ROWS * CDIM * 2);
  _Float16* wqkvT   = (_Float16*)carve((size_t)QKVN * CDIM * 2);
  _Float16* wprojT  = (_Float16*)carve((size_t)CDIM * CDIM * 2);
  _Float16* qkvbuf  = (_Float16*)carve((size_t)ROWS * QKVN * 2);
  _Float16* Qb      = (_Float16*)carve((size_t)ROWS * CDIM * 2);
  _Float16* Kb      = (_Float16*)carve((size_t)ROWS * CDIM * 2);
  _Float16* Vtb     = (_Float16*)carve((size_t)ROWS * CDIM * 2);
  _Float16* attn_o  = (_Float16*)carve((size_t)ROWS * CDIM * 2);

  // 0) weights -> f16 transposed
  {
    const int total = QKVN * CDIM + CDIM * CDIM;   // 2,359,296
    prep_weights_kernel<<<total / 256, 256, 0, stream>>>(w_qkv, w_proj, wqkvT, wprojT);
  }
  // 1) LayerNorm
  ln_kernel<<<ROWS, 256, 0, stream>>>(x, ln_g, ln_b, xn);
  // 2) QKV GEMM: [8192 x 2304] = xn * wqkvT^T   (64x64 tile per wave)
  gemm_f16_wmma<false><<<dim3(ROWS / 256, QKVN / 64), 128, 0, stream>>>(
      xn, wqkvT, b_qkv, qkvbuf, ROWS, QKVN, CDIM);
  // 3) RoPE + head-major repack (V transposed)
  rope_repack_kernel<<<(ROWS * CDIM) / 256, 256, 0, stream>>>(qkvbuf, Qb, Kb, Vtb);
  // 4) attention (32 q rows per wave)
  attn_kernel<<<dim3(NTOK / 128, NH, BTD), 128, 0, stream>>>(Qb, Kb, Vtb, attn_o);
  // 5) output projection -> f32 d_out
  gemm_f16_wmma<true><<<dim3(ROWS / 256, CDIM / 64), 128, 0, stream>>>(
      attn_o, wprojT, b_proj, out, ROWS, CDIM, CDIM);
}
